// BidirMambaBlock_51891794871146
// MI455X (gfx1250) — compile-verified, matrix operand order checked by reading, not run on hardware
//
#include <hip/hip_runtime.h>
#include <cstddef>

// Problem constants (match reference)
#define B_  4
#define L_  1024
#define D_  1024
#define DI_ 2048
#define DS_ 16
#define DC_ 4
#define DR_ 64
#define M_  (B_ * L_)   // 4096 token rows

typedef _Float16 v16h __attribute__((ext_vector_type(16)));
typedef _Float16 v8h  __attribute__((ext_vector_type(8)));
typedef _Float16 v4h  __attribute__((ext_vector_type(4)));
typedef float    v8f  __attribute__((ext_vector_type(8)));

enum { GF_REV_IN = 1, GF_REV_OUT = 2, GF_BIAS_SOFTPLUS = 4 };

__device__ __forceinline__ int rev_row(int r) {
    int b = r / L_;
    int t = r - b * L_;
    return b * L_ + (L_ - 1 - t);
}

// gfx1250 async copy: 16 bytes per lane, global -> LDS, tracked by ASYNCcnt.
__device__ __forceinline__ void async_copy_b128(unsigned lds_byte_addr, const void* gaddr) {
    asm volatile("global_load_async_to_lds_b128 %0, %1, off"
                 :: "v"(lds_byte_addr), "v"(gaddr) : "memory");
}
__device__ __forceinline__ void wait_async0() {
#if __has_builtin(__builtin_amdgcn_s_wait_asynccnt)
    __builtin_amdgcn_s_wait_asynccnt(0);
#else
    asm volatile("s_wait_asynccnt 0" ::: "memory");
#endif
}

// ---------------------------------------------------------------------------
// Tiled GEMM: C[M,N] = epilogue(A[M,K] @ W[N,K]^T), f16 operands in memory,
// f16 WMMA with f32 accumulation. BM=128 x BN=128 per 256-thread block
// (8 waves in 2x4 grid; each wave owns 64x32 = 4x2 v_wmma tiles).
// Staging: double-buffered LDS via global_load_async_to_lds_b128.
// ---------------------------------------------------------------------------
__global__ void __launch_bounds__(256)
gemm_wmma_f16(const _Float16* __restrict__ A, int lda,
              const _Float16* __restrict__ W,          // [N, K] row-major
              const float* __restrict__ bias,
              float* __restrict__ C, _Float16* __restrict__ C16, int ldc,
              int N, int K, int flags)
{
    constexpr int BM = 128, BN = 128, BK = 32;
    constexpr int PADH  = 40;          // halfs per LDS row (80B: 16B-aligned, bank-spread)
    constexpr int TILEH = BM * PADH;   // 5120 halfs = 10240 B per buffer
    __shared__ __align__(16) _Float16 As[2 * TILEH];
    __shared__ __align__(16) _Float16 Bs[2 * TILEH];

    const int tid  = threadIdx.x;
    const int wave = tid >> 5;
    const int lane = tid & 31;
    const int wm = wave & 1;        // 2 waves along M (64 rows each)
    const int wn = wave >> 1;       // 4 waves along N (32 cols each)
    const int fr = lane & 15;       // fragment row (M for A, N for B/D)
    const int hf = lane >> 4;       // lane half (K-group selector)

    const int blockM = blockIdx.y * BM;
    const int blockN = blockIdx.x * BN;

    // Zero-fill out-of-range B rows (async loads can't zero-fill); data cols only.
    const int nValid = N - blockN;           // rows >= nValid are OOB
    if (nValid < BN) {
        for (int i = tid; i < (BN - nValid) * BK; i += 256) {
            int r = nValid + i / BK;
            int c = i % BK;
            Bs[r * PADH + c]         = (_Float16)0.f;
            Bs[TILEH + r * PADH + c] = (_Float16)0.f;
        }
    }

    // Async stage of one K-tile into LDS buffer `buf`.
    auto stageA = [&](int buf, int k0) {
        #pragma unroll
        for (int i = 0; i < 2; ++i) {
            int idx = tid + 256 * i;         // 0..511 chunk id
            int row = idx >> 2;              // 0..127
            int ch  = idx & 3;               // 16B chunk within 64B row
            int gr  = blockM + row;
            int sr  = (flags & GF_REV_IN) ? rev_row(gr) : gr;
            unsigned lo = (unsigned)(size_t)&As[buf * TILEH + row * PADH + ch * 8];
            async_copy_b128(lo, A + (size_t)sr * lda + k0 + ch * 8);
        }
    };
    auto stageB = [&](int buf, int k0) {
        #pragma unroll
        for (int i = 0; i < 2; ++i) {
            int idx = tid + 256 * i;
            int row = idx >> 2;
            int ch  = idx & 3;
            int n   = blockN + row;
            if (n < N) {
                unsigned lo = (unsigned)(size_t)&Bs[buf * TILEH + row * PADH + ch * 8];
                async_copy_b128(lo, W + (size_t)n * K + k0 + ch * 8);
            }
        }
    };

    v8f acc[4][2] = {};

    stageA(0, 0);
    stageB(0, 0);
    wait_async0();
    __syncthreads();

    int buf = 0;
    for (int k0 = 0; k0 < K; k0 += BK) {
        const int nk = k0 + BK;
        if (nk < K) { stageA(buf ^ 1, nk); stageB(buf ^ 1, nk); }

        // Fragment loads per documented CDNA5 16-bit A/B layout:
        // lane (fr,hf): elems 0..7 = K[hf*8..+7], elems 8..15 = K[16+hf*8..+7]
        const _Float16* Ab = &As[buf * TILEH];
        const _Float16* Bb = &Bs[buf * TILEH];
        v16h afrag[4], bfrag[2];
        #pragma unroll
        for (int mt = 0; mt < 4; ++mt) {
            int ar = wm * 64 + mt * 16 + fr;
            v8h lo = *(const v8h*)&Ab[ar * PADH + hf * 8];
            v8h hi = *(const v8h*)&Ab[ar * PADH + 16 + hf * 8];
            #pragma unroll
            for (int e = 0; e < 8; ++e) { afrag[mt][e] = lo[e]; afrag[mt][8 + e] = hi[e]; }
        }
        #pragma unroll
        for (int nt = 0; nt < 2; ++nt) {
            int br = wn * 32 + nt * 16 + fr;
            v8h lo = *(const v8h*)&Bb[br * PADH + hf * 8];
            v8h hi = *(const v8h*)&Bb[br * PADH + 16 + hf * 8];
            #pragma unroll
            for (int e = 0; e < 8; ++e) { bfrag[nt][e] = lo[e]; bfrag[nt][8 + e] = hi[e]; }
        }

        #pragma unroll
        for (int mt = 0; mt < 4; ++mt)
            #pragma unroll
            for (int nt = 0; nt < 2; ++nt)
                acc[mt][nt] = __builtin_amdgcn_wmma_f32_16x16x32_f16(
                    false, afrag[mt], false, bfrag[nt],
                    (short)0, acc[mt][nt], false, false);

        if (nk < K) wait_async0();   // next tile landed in LDS
        __syncthreads();             // everyone done reading buf / writing buf^1
        buf ^= 1;
    }

    // Epilogue: D layout — VGPR r holds row (r + 8*hf), col = fr.
    #pragma unroll
    for (int mt = 0; mt < 4; ++mt) {
        #pragma unroll
        for (int nt = 0; nt < 2; ++nt) {
            int gcol = blockN + wn * 32 + nt * 16 + fr;
            if (gcol >= N) continue;
            #pragma unroll
            for (int r = 0; r < 8; ++r) {
                int grow = blockM + wm * 64 + mt * 16 + r + 8 * hf;
                float v = acc[mt][nt][r];
                if (flags & GF_BIAS_SOFTPLUS) {
                    v += bias[gcol];
                    v = (v > 20.f) ? v : log1pf(expf(v));
                }
                int orow = (flags & GF_REV_OUT) ? rev_row(grow) : grow;
                C[(size_t)orow * ldc + gcol] = v;
                if (C16) C16[(size_t)orow * ldc + gcol] = (_Float16)v;
            }
        }
    }
}

// ---------------------------------------------------------------------------
// fp32 -> f16 bulk conversion (vectorized x4)
// ---------------------------------------------------------------------------
__global__ void __launch_bounds__(256)
cvt_f32_f16(const float* __restrict__ s, _Float16* __restrict__ d, int n4)
{
    int i = blockIdx.x * 256 + threadIdx.x;
    if (i >= n4) return;
    float4 v = ((const float4*)s)[i];
    v4h o;
    o[0] = (_Float16)v.x; o[1] = (_Float16)v.y;
    o[2] = (_Float16)v.z; o[3] = (_Float16)v.w;
    ((v4h*)d)[i] = o;
}

// ---------------------------------------------------------------------------
// Causal depthwise conv1d (DC=4) + SiLU; emits f32 (scan) and f16 (GEMM) copies.
// ---------------------------------------------------------------------------
__global__ void __launch_bounds__(256)
conv_silu_kernel(const float* __restrict__ xz,   // [M_, 2*DI]
                 const float* __restrict__ cw,   // [DI, DC]
                 const float* __restrict__ cb,   // [DI]
                 float* __restrict__ xc, _Float16* __restrict__ xc16)
{
    int idx = blockIdx.x * 256 + threadIdx.x;
    if (idx >= M_ * DI_) return;
    int c = idx % DI_;
    int r = idx / DI_;
    int t = r % L_;
    float acc = cb[c];
    #pragma unroll
    for (int k = 0; k < DC_; ++k) {
        int ts = t - (DC_ - 1) + k;
        if (ts >= 0)
            acc += xz[(size_t)(r - (DC_ - 1) + k) * (2 * DI_) + c] * cw[c * DC_ + k];
    }
    acc = acc / (1.f + expf(-acc));       // silu
    xc[(size_t)r * DI_ + c]   = acc;
    xc16[(size_t)r * DI_ + c] = (_Float16)acc;
}

// ---------------------------------------------------------------------------
// Selective scan: one thread per (batch, channel), 16 states in registers.
// Fuses + u*D and silu(z) gate; emits f16 (only consumer is out_proj GEMM).
// ---------------------------------------------------------------------------
__global__ void __launch_bounds__(256)
scan_kernel(const float* __restrict__ xc,      // [M_, DI] u
            const float* __restrict__ delta,   // [M_, DI]
            const float* __restrict__ dbc,     // [M_, 96] (dt|B|C)
            const float* __restrict__ xz,      // [M_, 2*DI] (z = cols [DI,2*DI))
            const float* __restrict__ A_log,   // [DI, DS]
            const float* __restrict__ Dsk,     // [DI]
            _Float16* __restrict__ y16)        // [M_, DI]
{
    int idx = blockIdx.x * 256 + threadIdx.x;
    if (idx >= B_ * DI_) return;
    int b = idx / DI_;
    int c = idx % DI_;

    float Av[DS_];
    #pragma unroll
    for (int s = 0; s < DS_; ++s) Av[s] = -expf(A_log[c * DS_ + s]);
    float Dv = Dsk[c];

    float h[DS_] = {};
    for (int t = 0; t < L_; ++t) {
        size_t row = (size_t)b * L_ + t;
        float d = delta[row * DI_ + c];
        float u = xc[row * DI_ + c];
        const float* bc = dbc + row * (DR_ + 2 * DS_);
        float y = 0.f;
        #pragma unroll
        for (int s = 0; s < DS_; ++s) {
            float Bm = bc[DR_ + s];
            float Cm = bc[DR_ + DS_ + s];
            h[s] = h[s] * expf(d * Av[s]) + d * u * Bm;
            y += h[s] * Cm;
        }
        float z  = xz[row * (2 * DI_) + DI_ + c];
        float sz = z / (1.f + expf(-z));
        y16[row * DI_ + c] = (_Float16)((y + u * Dv) * sz);
    }
}

// ---------------------------------------------------------------------------
// Fused residual (x + yf + yb) + LayerNorm over D=1024. One block per row.
// ---------------------------------------------------------------------------
__global__ void __launch_bounds__(256)
ln_kernel(const float* __restrict__ x,
          const float* __restrict__ yf,
          const float* __restrict__ yb,
          const float* __restrict__ w,
          const float* __restrict__ bgain,
          float* __restrict__ out)
{
    __shared__ float red[256];
    const int row = blockIdx.x;
    const int tid = threadIdx.x;
    float vals[4];
    float s = 0.f;
    #pragma unroll
    for (int i = 0; i < 4; ++i) {
        int c = tid + 256 * i;
        size_t off = (size_t)row * D_ + c;
        float v = x[off] + yf[off] + yb[off];
        vals[i] = v; s += v;
    }
    red[tid] = s; __syncthreads();
    for (int o = 128; o > 0; o >>= 1) { if (tid < o) red[tid] += red[tid + o]; __syncthreads(); }
    float mu = red[0] * (1.f / D_);
    __syncthreads();
    float vs = 0.f;
    #pragma unroll
    for (int i = 0; i < 4; ++i) { float d = vals[i] - mu; vs += d * d; }
    red[tid] = vs; __syncthreads();
    for (int o = 128; o > 0; o >>= 1) { if (tid < o) red[tid] += red[tid + o]; __syncthreads(); }
    float inv = rsqrtf(red[0] * (1.f / D_) + 1e-5f);
    #pragma unroll
    for (int i = 0; i < 4; ++i) {
        int c = tid + 256 * i;
        out[(size_t)row * D_ + c] = (vals[i] - mu) * inv * w[c] + bgain[c];
    }
}

// ---------------------------------------------------------------------------
extern "C" void kernel_launch(void* const* d_in, const int* in_sizes, int n_in,
                              void* d_out, int out_size, void* d_ws, size_t ws_size,
                              hipStream_t stream)
{
    (void)in_sizes; (void)n_in; (void)out_size; (void)ws_size;

    const float* x = (const float*)d_in[0];

    // ---- workspace layout: fp32 region then f16 region ----
    float* ws    = (float*)d_ws;
    float* xz    = ws;                                  // [M_, 2*DI]
    float* xc    = xz    + (size_t)M_ * 2 * DI_;        // [M_, DI]
    float* delta = xc    + (size_t)M_ * DI_;            // [M_, DI]
    float* dbc   = delta + (size_t)M_ * DI_;            // [M_, 96]
    float* yf    = dbc   + (size_t)M_ * 96;             // [M_, D]
    float* yb    = yf    + (size_t)M_ * D_;             // [M_, D]
    _Float16* x16   = (_Float16*)(yb + (size_t)M_ * D_);
    _Float16* xc16  = x16   + (size_t)M_ * D_;          // [M_, DI]
    _Float16* dbc16 = xc16  + (size_t)M_ * DI_;         // [M_, 96]
    _Float16* y16   = dbc16 + (size_t)M_ * 96;          // [M_, DI]
    _Float16* w16   = y16   + (size_t)M_ * DI_;         // scratch, up to 4M halfs

    const dim3 blk(256);
    auto cvt = [&](const float* s, _Float16* d, int n) {
        cvt_f32_f16<<<(n / 4 + 255) / 256, blk, 0, stream>>>(s, d, n / 4);
    };

    // x -> f16 once (shared by both directions; backward flips at GEMM load)
    cvt(x, x16, M_ * D_);

    for (int p = 0; p < 2; ++p) {
        const int o = 1 + p * 9;
        const float* in_proj  = (const float*)d_in[o + 0]; // [2*DI, D]
        const float* conv_w   = (const float*)d_in[o + 1]; // [DI, DC]
        const float* conv_b   = (const float*)d_in[o + 2]; // [DI]
        const float* x_proj   = (const float*)d_in[o + 3]; // [96, DI]
        const float* dt_w     = (const float*)d_in[o + 4]; // [DI, DR]
        const float* dt_b     = (const float*)d_in[o + 5]; // [DI]
        const float* A_log    = (const float*)d_in[o + 6]; // [DI, DS]
        const float* Dw       = (const float*)d_in[o + 7]; // [DI]
        const float* out_proj = (const float*)d_in[o + 8]; // [D, DI]
        float* ydir = (p == 0) ? yf : yb;
        const int rin  = p ? GF_REV_IN  : 0;
        const int rout = p ? GF_REV_OUT : 0;

        // 1. xz = x @ in_proj^T  (N=4096, K=1024); backward: time-flip A rows
        cvt(in_proj, w16, 2 * DI_ * D_);
        gemm_wmma_f16<<<dim3((2 * DI_ + 127) / 128, M_ / 128), blk, 0, stream>>>(
            x16, D_, w16, nullptr, xz, nullptr, 2 * DI_, 2 * DI_, D_, rin);

        // 2. depthwise causal conv + silu
        conv_silu_kernel<<<(M_ * DI_ + 255) / 256, blk, 0, stream>>>(
            xz, conv_w, conv_b, xc, xc16);

        // 3. dbc = xc @ x_proj^T  (N=96, K=2048); also emit f16 copy for dt GEMM
        cvt(x_proj, w16, (DR_ + 2 * DS_) * DI_);
        gemm_wmma_f16<<<dim3(1, M_ / 128), blk, 0, stream>>>(
            xc16, DI_, w16, nullptr, dbc, dbc16, DR_ + 2 * DS_,
            DR_ + 2 * DS_, DI_, 0);

        // 4. delta = softplus(dt @ dt_w^T + dt_b)  (N=2048, K=64, A = dbc16[:, :64])
        cvt(dt_w, w16, DI_ * DR_);
        gemm_wmma_f16<<<dim3(DI_ / 128, M_ / 128), blk, 0, stream>>>(
            dbc16, DR_ + 2 * DS_, w16, dt_b, delta, nullptr, DI_,
            DI_, DR_, GF_BIAS_SOFTPLUS);

        // 5. selective scan (+ u*D, * silu(z)) -> f16
        scan_kernel<<<(B_ * DI_ + 255) / 256, blk, 0, stream>>>(
            xc, delta, dbc, xz, A_log, Dw, y16);

        // 6. ydir = y @ out_proj^T  (N=1024, K=2048); backward: time-flip output rows
        cvt(out_proj, w16, D_ * DI_);
        gemm_wmma_f16<<<dim3(D_ / 128, M_ / 128), blk, 0, stream>>>(
            y16, DI_, w16, nullptr, ydir, nullptr, D_, D_, DI_, rout);
    }

    const float* ln_w = (const float*)d_in[19];
    const float* ln_b = (const float*)d_in[20];
    ln_kernel<<<M_, blk, 0, stream>>>(x, yf, yb, ln_w, ln_b, (float*)d_out);
}
